// mLSTMblock_5557687681258
// MI455X (gfx1250) — compile-verified, matrix operand order checked by reading, not run on hardware
//
#include <hip/hip_runtime.h>

typedef __attribute__((ext_vector_type(16))) _Float16 v16h;
typedef __attribute__((ext_vector_type(8)))  float    v8f;

#define B_   4
#define T_   256
#define D_   256
#define D2_  512
#define H_   4
#define DH_  128
#define TP_  (T_ + 3)
#define SCALE_ 0.0625f   // D^-0.5 = 256^-0.5

#if defined(__gfx1250__) && __has_builtin(__builtin_amdgcn_tensor_load_to_lds) && \
    __has_builtin(__builtin_amdgcn_s_wait_tensorcnt)
#define USE_TDM 1
typedef unsigned int u32x4 __attribute__((ext_vector_type(4)));
typedef int          i32x4 __attribute__((ext_vector_type(4)));
typedef int          i32x8 __attribute__((ext_vector_type(8)));

// TDM 1-D row load: 512 x f32 (2KB) global -> LDS. D# layout per CDNA5 ISA §8.
// amdgpu-toolchain (clang-23) 6-arg builtin form.
__device__ __forceinline__ void tdm_load_row(unsigned lds_off, const float* gptr) {
  unsigned long long ga = (unsigned long long)(uintptr_t)gptr;
  u32x4 g0;
  g0[0] = 1u;                                            // count=1, user mode
  g0[1] = lds_off;                                       // lds_addr (bytes)
  g0[2] = (unsigned)(ga & 0xFFFFFFFFu);                  // global_addr[31:0]
  g0[3] = (unsigned)((ga >> 32) & 0x1FFFFFFu)            // global_addr[56:32]
        | (2u << 30);                                    // type=2 ("image")
  i32x8 g1;
  g1[0] = (2 << 16);                                     // data_size=2 (4B), mask=0
  g1[1] = (int)((512u & 0xFFFFu) << 16);                 // tensor_dim0[15:0] @63:48
  g1[2] = 0;                                             // tensor_dim0 hi, dim1 lo
  g1[3] = (int)(512u << 16);                             // tile_dim0=512 @127:112
  g1[4] = 0;                                             // tile_dim1/2 = 0
  g1[5] = 512;                                           // tensor_dim0_stride
  g1[6] = 0;
  g1[7] = 0;
  i32x4 z4 = {};
  i32x8 z8 = {};
  __builtin_amdgcn_tensor_load_to_lds(g0, g1, z4, z4, z8, 0);
}
#else
#define USE_TDM 0
#endif

// ---------------------------------------------------------------------------
// LayerNorm over D=256, writes f16 x_norm. One block (256 thr) per (b,t) row.
// ---------------------------------------------------------------------------
__global__ void k_layernorm(const float* __restrict__ x, const float* __restrict__ g,
                            const float* __restrict__ b, _Float16* __restrict__ xn) {
  __shared__ float s1[256], s2[256];
  int r = blockIdx.x, t = threadIdx.x;
  float v = x[(size_t)r * D_ + t];
  s1[t] = v; s2[t] = v * v; __syncthreads();
  for (int s = 128; s > 0; s >>= 1) {
    if (t < s) { s1[t] += s1[t + s]; s2[t] += s2[t + s]; }
    __syncthreads();
  }
  float mu  = s1[0] * (1.0f / D_);
  float var = s2[0] * (1.0f / D_) - mu * mu;
  float rs  = rsqrtf(var + 1e-5f);
  xn[(size_t)r * D_ + t] = (_Float16)((v - mu) * rs * g[t] + b[t]);
}

// ---------------------------------------------------------------------------
__global__ void k_cvt16(const float* __restrict__ s, _Float16* __restrict__ d, int n) {
  int i = blockIdx.x * blockDim.x + threadIdx.x;
  if (i < n) d[i] = (_Float16)s[i];
}

// conv_w (D2,D2,4) [o][i][j] -> Wc_h [o][j*D2 + i] (im2col weight layout), f16
__global__ void k_conv_w(const float* __restrict__ s, _Float16* __restrict__ d) {
  int i = blockIdx.x * blockDim.x + threadIdx.x;
  if (i < D2_ * 4 * D2_) {
    int o = i / (4 * D2_); int rem = i % (4 * D2_);
    int j = rem / D2_;     int ci  = rem % D2_;
    d[i] = (_Float16)s[((size_t)o * D2_ + ci) * 4 + j];
  }
}

// x_left (B,T,D2) f32 -> padded f16 (B,T+3,D2) with 3 zero rows per batch
__global__ void k_pad_xl(const float* __restrict__ xl, _Float16* __restrict__ xp) {
  int i = blockIdx.x * blockDim.x + threadIdx.x;
  if (i < B_ * TP_ * D2_) {
    int b = i / (TP_ * D2_); int rem = i % (TP_ * D2_);
    int t = rem / D2_;       int c   = rem % D2_;
    xp[i] = (t < 3) ? (_Float16)0.0f
                    : (_Float16)xl[((size_t)b * T_ + (t - 3)) * D2_ + c];
  }
}

// ---------------------------------------------------------------------------
// i_bar / f_bar GEMV: one block (128 thr) per row of x_left
// ---------------------------------------------------------------------------
__global__ void k_if_proj(const float* __restrict__ xl,
                          const float* __restrict__ Wi, const float* __restrict__ bi,
                          const float* __restrict__ Wf, const float* __restrict__ bf,
                          float* __restrict__ ib, float* __restrict__ fb) {
  __shared__ float si[128], sf[128];
  int r = blockIdx.x, t = threadIdx.x;
  float ai = 0.0f, af = 0.0f;
  for (int c = t; c < D2_; c += 128) {
    float xv = xl[(size_t)r * D2_ + c];
    ai += xv * Wi[c]; af += xv * Wf[c];
  }
  si[t] = ai; sf[t] = af; __syncthreads();
  for (int s = 64; s > 0; s >>= 1) {
    if (t < s) { si[t] += si[t + s]; sf[t] += sf[t + s]; }
    __syncthreads();
  }
  if (t == 0) { ib[r] = si[0] + bi[0]; fb[r] = sf[0] + bf[0]; }
}

// ---------------------------------------------------------------------------
// WMMA GEMM, register-blocked: each wave computes a 16x64 C tile (4 acc tiles,
// A fragment reused 4x).  C(M,N) = act(A@W^T * scale + bias) + res.
// A rows may live in a padded buffer: physical row = r + pad*(r/Tr + 1)
// (tiles never straddle batch boundaries since T % 16 == 0).
// act: 0=none, 1=sigmoid, 2=swish.  Layouts per CDNA5 ISA 7.12.2.
// ---------------------------------------------------------------------------
__global__ __launch_bounds__(32)
void k_wmma_gemm(const _Float16* __restrict__ A, int lda, int Tr, int pad,
                 const _Float16* __restrict__ W, int ldw,
                 float* __restrict__ C, int ldc, int K,
                 const float* __restrict__ bias, float scale, int act,
                 const float* __restrict__ res) {
  int n0 = blockIdx.x * 64, m0 = blockIdx.y * 16;
  int lane = threadIdx.x & 31;
  int lm = lane & 15, hi = lane >> 4;
  const _Float16* Ap = A + (size_t)(m0 + pad * (m0 / Tr + 1) + lm) * lda;
  const _Float16* Wp = W + (size_t)(n0 + lm) * ldw;
  v8f accs[4] = {};
  for (int k0 = 0; k0 < K; k0 += 32) {
    if (k0 + 32 < K) {  // prefetch next K-slab into near caches
      __builtin_prefetch(Ap + k0 + 32, 0, 3);
      __builtin_prefetch(Wp + k0 + 32, 0, 3);
    }
    v16h af, b0, b1, b2, b3;
    #pragma unroll
    for (int e = 0; e < 16; e++) {
      int v = e >> 1, p = e & 1;
      int kk = ((v >> 2) << 4) + (hi << 3) + ((v & 3) << 1) + p + k0;
      af[e] = Ap[kk];
      b0[e] = Wp[kk];
      b1[e] = Wp[(size_t)16 * ldw + kk];
      b2[e] = Wp[(size_t)32 * ldw + kk];
      b3[e] = Wp[(size_t)48 * ldw + kk];
    }
    accs[0] = __builtin_amdgcn_wmma_f32_16x16x32_f16(false, af, false, b0,
                                                     (short)0, accs[0], false, false);
    accs[1] = __builtin_amdgcn_wmma_f32_16x16x32_f16(false, af, false, b1,
                                                     (short)0, accs[1], false, false);
    accs[2] = __builtin_amdgcn_wmma_f32_16x16x32_f16(false, af, false, b2,
                                                     (short)0, accs[2], false, false);
    accs[3] = __builtin_amdgcn_wmma_f32_16x16x32_f16(false, af, false, b3,
                                                     (short)0, accs[3], false, false);
  }
  #pragma unroll
  for (int nt = 0; nt < 4; nt++) {
    #pragma unroll
    for (int r = 0; r < 8; r++) {
      int m = m0 + r + 8 * hi, n = n0 + nt * 16 + lm;
      float v = accs[nt][r] * scale;
      if (bias) v += bias[n];
      if (act == 1)      v = 1.0f / (1.0f + __expf(-v));
      else if (act == 2) v = v / (1.0f + __expf(-v));
      size_t idx = (size_t)m * ldc + n;
      if (res) v += res[idx];
      C[idx] = v;
    }
  }
}

// ---------------------------------------------------------------------------
// mLSTM scan. Each workgroup owns a 16-row slice of c (16x512 f32, 32KB LDS)
// for one batch; 256 threads = 16 rows x 16 col-segments of 32 cols.
// Per-step k/q rows staged to LDS via the Tensor Data Mover (double-buffered,
// issued one step ahead, synchronized with s_wait_tensorcnt) when available.
// ---------------------------------------------------------------------------
#define ROWS_ 16
__global__ __launch_bounds__(256)
void k_scan(const float* __restrict__ ib, const float* __restrict__ fb,
            const float* __restrict__ q,  const float* __restrict__ k,
            const float* __restrict__ v,  const float* __restrict__ o,
            float* __restrict__ hseq) {
  __shared__ float c[ROWS_][D2_];
  __shared__ float kq[2][2][D2_];     // [buf][k=0/q=1][col]
  __shared__ float nv[D2_];
  __shared__ float red[256], cqv[ROWS_];
  __shared__ float m_sh, den_sh;
  int b  = blockIdx.x / (D2_ / ROWS_);
  int r0 = (blockIdx.x % (D2_ / ROWS_)) * ROWS_;
  int tid = threadIdx.x;
  int row = tid >> 4, seg = tid & 15;
  for (int i = tid; i < ROWS_ * D2_; i += 256) ((float*)c)[i] = 0.0f;
  nv[tid * 2] = 0.0f; nv[tid * 2 + 1] = 0.0f;
  if (tid == 0) m_sh = 0.0f;
#if USE_TDM
  int wave = tid >> 5;
  unsigned kq_lds = (unsigned)(uintptr_t)&kq[0][0][0];
  if (wave == 0) {  // prologue: stage t=0 into buffer 0
    size_t base0 = (size_t)(b * T_) * D2_;
    tdm_load_row(kq_lds,               k + base0);
    tdm_load_row(kq_lds + D2_ * 4,     q + base0);
  }
#endif
  __syncthreads();
  for (int t = 0; t < T_; t++) {
    size_t base = (size_t)(b * T_ + t) * D2_;
    int buf = t & 1;
#if USE_TDM
    if (wave == 0) __builtin_amdgcn_s_wait_tensorcnt((short)0);
    __syncthreads();                   // staged k/q visible to all waves
    if (wave == 0 && t + 1 < T_) {     // issue next step into other buffer
      size_t basen = base + D2_;
      unsigned boff = kq_lds + (unsigned)((buf ^ 1) * 2 * D2_ * 4);
      tdm_load_row(boff,            k + basen);
      tdm_load_row(boff + D2_ * 4,  q + basen);
    }
#else
    kq[buf][0][tid * 2]     = k[base + tid * 2];
    kq[buf][0][tid * 2 + 1] = k[base + tid * 2 + 1];
    kq[buf][1][tid * 2]     = q[base + tid * 2];
    kq[buf][1][tid * 2 + 1] = q[base + tid * 2 + 1];
    __syncthreads();
#endif
    float ibv = ib[b * T_ + t], fbv = fb[b * T_ + t];
    float mp = m_sh;
    float mn = fmaxf(fbv + mp, ibv);
    float ig = __expf(ibv - mn);
    float fg = __expf(fbv + mp - mn);
    float vr = v[base + r0 + row];
    float part = 0.0f;
    int c0 = seg * 32;
    #pragma unroll 8
    for (int j = 0; j < 32; j++) {
      int col = c0 + j;
      float cc = fg * c[row][col] + ig * vr * kq[buf][0][col];
      c[row][col] = cc;
      part += cc * kq[buf][1][col];
    }
    red[tid] = part; __syncthreads();
    for (int s = 8; s > 0; s >>= 1) {
      if (seg < s) red[tid] += red[tid + s];
      __syncthreads();
    }
    if (seg == 0) cqv[row] = red[tid];
    float dp;
    {
      int c2 = tid * 2;
      float n0v = fg * nv[c2]     + ig * kq[buf][0][c2];     nv[c2]     = n0v;
      float n1v = fg * nv[c2 + 1] + ig * kq[buf][0][c2 + 1]; nv[c2 + 1] = n1v;
      dp = n0v * kq[buf][1][c2] + n1v * kq[buf][1][c2 + 1];
    }
    __syncthreads();
    red[tid] = dp; __syncthreads();
    for (int s = 128; s > 0; s >>= 1) {
      if (tid < s) red[tid] += red[tid + s];
      __syncthreads();
    }
    if (tid == 0) {
      den_sh = fmaxf(fabsf(red[0]), 1.0f) + 1e-8f;
      m_sh   = mn;
    }
    __syncthreads();
    if (tid < ROWS_) {
      int d = r0 + tid;
      hseq[base + d] = o[base + d] * cqv[tid] / den_sh;
    }
    __syncthreads();
  }
}

// ---------------------------------------------------------------------------
// GroupNorm per (b, head) over (T, DH) + skip + swish(x_right) gate -> f16 h
// ---------------------------------------------------------------------------
__global__ void k_groupnorm(const float* __restrict__ hseq, const float* __restrict__ skip,
                            const float* __restrict__ xr,   const float* __restrict__ gg,
                            const float* __restrict__ gb,   _Float16* __restrict__ hh) {
  __shared__ float s1[256], s2[256];
  int b = blockIdx.x / H_, h = blockIdx.x % H_;
  int tid = threadIdx.x;
  float a = 0.0f, a2 = 0.0f;
  for (int i = tid; i < T_ * DH_; i += 256) {
    int t = i >> 7, dh = i & (DH_ - 1);
    float vv = hseq[(size_t)(b * T_ + t) * D2_ + h * DH_ + dh];
    a += vv; a2 += vv * vv;
  }
  s1[tid] = a; s2[tid] = a2; __syncthreads();
  for (int s = 128; s > 0; s >>= 1) {
    if (tid < s) { s1[tid] += s1[tid + s]; s2[tid] += s2[tid + s]; }
    __syncthreads();
  }
  float mu  = s1[0] / (float)(T_ * DH_);
  float var = s2[0] / (float)(T_ * DH_) - mu * mu;
  float rs  = rsqrtf(var + 1e-5f);
  for (int i = tid; i < T_ * DH_; i += 256) {
    int t = i >> 7, dh = i & (DH_ - 1);
    int cch = h * DH_ + dh;
    size_t idx = (size_t)(b * T_ + t) * D2_ + cch;
    float zn = (hseq[idx] - mu) * rs;
    float hg = zn * gg[cch] + gb[cch] + skip[idx];
    float xv = xr[idx];
    float sw = xv / (1.0f + __expf(-xv));
    hh[idx]  = (_Float16)(hg * sw);
  }
}

// ---------------------------------------------------------------------------
extern "C" void kernel_launch(void* const* d_in, const int* in_sizes, int n_in,
                              void* d_out, int out_size, void* d_ws, size_t ws_size,
                              hipStream_t stream) {
  (void)in_sizes; (void)n_in; (void)out_size; (void)ws_size;
  const float* x      = (const float*)d_in[0];
  const float* ln_g   = (const float*)d_in[1];
  const float* ln_b   = (const float*)d_in[2];
  const float* W_left = (const float*)d_in[3];
  const float* b_left = (const float*)d_in[4];
  const float* W_right= (const float*)d_in[5];
  const float* b_right= (const float*)d_in[6];
  const float* Wi     = (const float*)d_in[7];
  const float* bi     = (const float*)d_in[8];
  const float* Wf     = (const float*)d_in[9];
  const float* bf     = (const float*)d_in[10];
  const float* Wo     = (const float*)d_in[11];
  const float* bo     = (const float*)d_in[12];
  const float* Wq     = (const float*)d_in[13];
  const float* Wk     = (const float*)d_in[14];
  const float* Wv     = (const float*)d_in[15];
  const float* conv_w = (const float*)d_in[16];
  const float* conv_b = (const float*)d_in[17];
  const float* skip_W = (const float*)d_in[18];
  const float* gn_g   = (const float*)d_in[19];
  const float* gn_b   = (const float*)d_in[20];
  const float* W_last = (const float*)d_in[21];
  const float* b_last = (const float*)d_in[22];
  float* out = (float*)d_out;

  char* ws = (char*)d_ws;
  size_t off = 0;
  auto alloc = [&](size_t bytes) -> char* {
    char* p = ws + off; off += (bytes + 255) & ~(size_t)255; return p;
  };
  const int M = B_ * T_;  // 1024

  _Float16* xn_h  = (_Float16*)alloc((size_t)M * D_ * 2);
  _Float16* wl_h  = (_Float16*)alloc((size_t)D2_ * D_ * 2);
  _Float16* wr_h  = (_Float16*)alloc((size_t)D2_ * D_ * 2);
  _Float16* wo_h  = (_Float16*)alloc((size_t)D2_ * D2_ * 2);
  _Float16* wc_h  = (_Float16*)alloc((size_t)D2_ * 4 * D2_ * 2);
  _Float16* wq_h  = (_Float16*)alloc((size_t)H_ * DH_ * DH_ * 2);
  _Float16* wk_h  = (_Float16*)alloc((size_t)H_ * DH_ * DH_ * 2);
  _Float16* wv_h  = (_Float16*)alloc((size_t)H_ * DH_ * DH_ * 2);
  _Float16* sk_h  = (_Float16*)alloc((size_t)D2_ * D2_ * 2);
  _Float16* wlt_h = (_Float16*)alloc((size_t)D_ * D2_ * 2);
  float* xl_f   = (float*)alloc((size_t)M * D2_ * 4);
  float* xr_f   = (float*)alloc((size_t)M * D2_ * 4);
  float* o_f    = (float*)alloc((size_t)M * D2_ * 4);
  float* xt_f   = (float*)alloc((size_t)M * D2_ * 4);
  float* q_f    = (float*)alloc((size_t)M * D2_ * 4);
  float* k_f    = (float*)alloc((size_t)M * D2_ * 4);
  float* v_f    = (float*)alloc((size_t)M * D2_ * 4);
  float* skip_f = (float*)alloc((size_t)M * D2_ * 4);
  float* hseq_f = (float*)alloc((size_t)M * D2_ * 4);
  _Float16* xlp_h = (_Float16*)alloc((size_t)B_ * TP_ * D2_ * 2);
  _Float16* xt_h  = (_Float16*)alloc((size_t)M * D2_ * 2);
  _Float16* hh_h  = (_Float16*)alloc((size_t)M * D2_ * 2);
  float* ib_f = (float*)alloc((size_t)M * 4);
  float* fb_f = (float*)alloc((size_t)M * 4);

  const int thr = 256;
  #define GS(n) dim3(((n) + thr - 1) / thr)

  // weight conversions
  k_cvt16<<<GS(D2_*D_), thr, 0, stream>>>(W_left,  wl_h,  D2_*D_);
  k_cvt16<<<GS(D2_*D_), thr, 0, stream>>>(W_right, wr_h,  D2_*D_);
  k_cvt16<<<GS(D2_*D2_), thr, 0, stream>>>(Wo,     wo_h,  D2_*D2_);
  k_cvt16<<<GS(H_*DH_*DH_), thr, 0, stream>>>(Wq,  wq_h,  H_*DH_*DH_);
  k_cvt16<<<GS(H_*DH_*DH_), thr, 0, stream>>>(Wk,  wk_h,  H_*DH_*DH_);
  k_cvt16<<<GS(H_*DH_*DH_), thr, 0, stream>>>(Wv,  wv_h,  H_*DH_*DH_);
  k_cvt16<<<GS(D2_*D2_), thr, 0, stream>>>(skip_W, sk_h,  D2_*D2_);
  k_cvt16<<<GS(D_*D2_), thr, 0, stream>>>(W_last,  wlt_h, D_*D2_);
  k_conv_w<<<GS(D2_*4*D2_), thr, 0, stream>>>(conv_w, wc_h);

  // LayerNorm
  k_layernorm<<<dim3(M), dim3(256), 0, stream>>>(x, ln_g, ln_b, xn_h);

  // x_left / x_right GEMMs: (1024x256) @ (512x256)^T  (N tiles of 64)
  k_wmma_gemm<<<dim3(D2_/64, M/16), dim3(32), 0, stream>>>(
      xn_h, D_, M, 0, wl_h, D_, xl_f, D2_, D_, b_left, 1.0f, 0, nullptr);
  k_wmma_gemm<<<dim3(D2_/64, M/16), dim3(32), 0, stream>>>(
      xn_h, D_, M, 0, wr_h, D_, xr_f, D2_, D_, b_right, 1.0f, 0, nullptr);

  // padded f16 x_left (also serves as plain f16 x_left via row remap)
  k_pad_xl<<<GS(B_*TP_*D2_), thr, 0, stream>>>(xl_f, xlp_h);

  // i_bar / f_bar
  k_if_proj<<<dim3(M), dim3(128), 0, stream>>>(xl_f, Wi, bi, Wf, bf, ib_f, fb_f);

  // o = sigmoid(x_left @ Wo^T + bo)
  k_wmma_gemm<<<dim3(D2_/64, M/16), dim3(32), 0, stream>>>(
      xlp_h, D2_, T_, 3, wo_h, D2_, o_f, D2_, D2_, bo, 1.0f, 1, nullptr);

  // conv as implicit-im2col GEMM, K = 4*512; swish epilogue
  k_wmma_gemm<<<dim3(D2_/64, M/16), dim3(32), 0, stream>>>(
      xlp_h - 3*D2_, D2_, T_, 3, wc_h, 4*D2_, xt_f, D2_, 4*D2_, conv_b, 1.0f, 2, nullptr);
  k_cvt16<<<GS(M*D2_), thr, 0, stream>>>(xt_f, xt_h, M*D2_);

  // block-diagonal q / k / v projections (per head)
  for (int h = 0; h < H_; h++) {
    k_wmma_gemm<<<dim3(DH_/64, M/16), dim3(32), 0, stream>>>(
        xt_h + h*DH_, D2_, M, 0, wq_h + h*DH_*DH_, DH_, q_f + h*DH_, D2_,
        DH_, nullptr, 1.0f, 0, nullptr);
    k_wmma_gemm<<<dim3(DH_/64, M/16), dim3(32), 0, stream>>>(
        xt_h + h*DH_, D2_, M, 0, wk_h + h*DH_*DH_, DH_, k_f + h*DH_, D2_,
        DH_, nullptr, SCALE_, 0, nullptr);
    k_wmma_gemm<<<dim3(DH_/64, M/16), dim3(32), 0, stream>>>(
        xlp_h + h*DH_, D2_, T_, 3, wv_h + h*DH_*DH_, DH_, v_f + h*DH_, D2_,
        DH_, nullptr, 1.0f, 0, nullptr);
  }

  // skip = x_trans @ skip_W^T
  k_wmma_gemm<<<dim3(D2_/64, M/16), dim3(32), 0, stream>>>(
      xt_h, D2_, M, 0, sk_h, D2_, skip_f, D2_, D2_, nullptr, 1.0f, 0, nullptr);

  // sequential mLSTM scan, 128 workgroups (B x 32 row-slices of c)
  k_scan<<<dim3(B_ * (D2_/ROWS_)), dim3(256), 0, stream>>>(
      ib_f, fb_f, q_f, k_f, v_f, o_f, hseq_f);

  // GroupNorm + skip + swish(x_right) gate -> f16 h
  k_groupnorm<<<dim3(B_*H_), dim3(256), 0, stream>>>(
      hseq_f, skip_f, xr_f, gn_g, gn_b, hh_h);

  // out = h @ W_last^T + b_last + x
  k_wmma_gemm<<<dim3(D_/64, M/16), dim3(32), 0, stream>>>(
      hh_h, D2_, M, 0, wlt_h, D2_, out, D_, D2_, b_last, 1.0f, 0, x);
  #undef GS
}
